// MultiHeadAttention_50800873177468
// MI455X (gfx1250) — compile-verified, matrix operand order checked by reading, not run on hardware
//
#include <hip/hip_runtime.h>

// ---------------------------------------------------------------------------
// MI455X (gfx1250) fused causal multi-head attention.
//   B=8, T=1024, D=1024, H=16, hd=64.  Outputs: out[B,T,D] fp32, qk[B,H,T,T] fp32.
// bf16 V_WMMA_F32_16X16X32_BF16 everywhere with fp32 accumulation; flash-style
// fused softmax*V writing qk exactly once; LDS-staged operands via
// GLOBAL_LOAD_ASYNC_TO_LDS_B128 (ASYNCcnt) with sync fallback.
// ---------------------------------------------------------------------------

typedef __attribute__((ext_vector_type(16))) __bf16 v16bf;
typedef __attribute__((ext_vector_type(8)))  float  v8f;
typedef unsigned short ushort_t;
typedef int v4i __attribute__((vector_size(16)));
typedef __attribute__((address_space(1))) v4i gv4i;   // global
typedef __attribute__((address_space(3))) v4i lv4i;   // LDS

struct __align__(16) U128 { unsigned int w[4]; };
struct __align__(16) F4   { float f[4]; };

union FragAB { v16bf v; U128 u[2]; };           // 32B WMMA 16-bit operand
union Pack8  { ushort_t h[8]; U128 u; };

#define NEG_INF (-__builtin_inff())

#if defined(__AMDGCN__) && __has_builtin(__builtin_amdgcn_global_load_async_to_lds_b128)
#define HAVE_ASYNC_LDS 1
#else
#define HAVE_ASYNC_LDS 0
#endif

// Stage one 16B chunk global -> LDS (per-lane addresses).
static __device__ inline void stage16(ushort_t* ldst, const ushort_t* gsrc) {
#if HAVE_ASYNC_LDS
    __builtin_amdgcn_global_load_async_to_lds_b128(
        (gv4i*)gsrc, (lv4i*)ldst, 0, 0);
#else
    *(U128*)ldst = *(const U128*)gsrc;
#endif
}

static __device__ inline void stage_wait() {
#if HAVE_ASYNC_LDS && __has_builtin(__builtin_amdgcn_s_wait_asynccnt)
    __builtin_amdgcn_s_wait_asynccnt(0);
#endif
    __syncthreads();
}

static __device__ inline ushort_t f2bf(float f) {
    unsigned u = __builtin_bit_cast(unsigned, f);
    u += 0x7FFFu + ((u >> 16) & 1u);            // round-to-nearest-even
    return (ushort_t)(u >> 16);
}

static __device__ inline v8f wmma_bf16(const FragAB& a, const FragAB& b, v8f c) {
    return __builtin_amdgcn_wmma_f32_16x16x32_bf16(
        false, a.v, false, b.v, (short)0, c, false, false);
}

// ---------------------------------------------------------------------------
// fp32 -> bf16 conversion (grid-stride)
// ---------------------------------------------------------------------------
__global__ __launch_bounds__(256) void cvt_bf16(const float* __restrict__ in,
                                                ushort_t* __restrict__ out, int n) {
    for (int i = blockIdx.x * blockDim.x + threadIdx.x; i < n;
         i += gridDim.x * blockDim.x)
        out[i] = f2bf(in[i]);
}

// ---------------------------------------------------------------------------
// Tiled GEMM:  C[M,N] = A[M,K] @ W[N,K]^T   (A, W bf16 row-major, K = 1024)
// Block tile 128x128, BK=32, 8 waves (4x2), wave tile 32x64 = 8 WMMA accums.
// mode 0: qh  = (C + bq)*scale  -> bf16 [B,H,T,hd]
// mode 1: kh  =  C      *scale  -> bf16 [B,H,T,hd]
// mode 2: vT  =  C + bv         -> bf16 [B,H,hd,T]  (transposed, packed stores)
// mode 3: out =  C + bo         -> fp32 [M,N]
// ---------------------------------------------------------------------------
__global__ __launch_bounds__(256) void gemm_bf16(
    const ushort_t* __restrict__ A, const ushort_t* __restrict__ W,
    const float* __restrict__ bias, void* __restrict__ outp,
    int mode, float scale) {

    __shared__ ushort_t As[128 * 40];   // stride 40 bf16 = 80B = 20 banks (conflict-free)
    __shared__ ushort_t Bs[128 * 40];

    const int tid  = threadIdx.x;
    const int wid  = tid >> 5, lane = tid & 31;
    const int wm   = wid & 3,  wn   = wid >> 2;       // 4x2 wave grid
    const int half = lane >> 4, l16 = lane & 15;
    const int bm   = blockIdx.y * 128, bn = blockIdx.x * 128;

    v8f acc[2][4] = {};

    const int lr = tid >> 2;              // 0..63
    const int lc = (tid & 3) * 8;         // 0,8,16,24 (bf16 elems = 16B chunks)

    for (int k0 = 0; k0 < 1024; k0 += 32) {
        // ---- stage 128x32 A and 128x32 B tiles into LDS ----
        #pragma unroll
        for (int p = 0; p < 2; ++p) {
            int row = lr + p * 64;
            stage16(&As[row * 40 + lc],
                    &A[(size_t)(bm + row) * 1024 + k0 + lc]);
            stage16(&Bs[row * 40 + lc],
                    &W[(size_t)(bn + row) * 1024 + k0 + lc]);
        }
        if (k0 + 32 < 1024) {             // L2 prefetch of next K-step
            __builtin_prefetch(&A[(size_t)(bm + lr) * 1024 + k0 + 32], 0, 1);
            __builtin_prefetch(&W[(size_t)(bn + lr) * 1024 + k0 + 32], 0, 1);
        }
        stage_wait();

        // ---- fragments: lane = M/N mod 16; K chunks at half*8 and half*8+16 ----
        FragAB af[2], bf[4];
        #pragma unroll
        for (int i = 0; i < 2; ++i) {
            int ra = wm * 32 + i * 16 + l16;
            af[i].u[0] = *(const U128*)&As[ra * 40 + half * 8];
            af[i].u[1] = *(const U128*)&As[ra * 40 + half * 8 + 16];
        }
        #pragma unroll
        for (int j = 0; j < 4; ++j) {
            int rb = wn * 64 + j * 16 + l16;
            bf[j].u[0] = *(const U128*)&Bs[rb * 40 + half * 8];
            bf[j].u[1] = *(const U128*)&Bs[rb * 40 + half * 8 + 16];
        }
        #pragma unroll
        for (int i = 0; i < 2; ++i)
            #pragma unroll
            for (int j = 0; j < 4; ++j)
                acc[i][j] = wmma_bf16(af[i], bf[j], acc[i][j]);
        __syncthreads();
    }

    // ---- epilogue; C layout: row = r + 8*half, col = lane&15 ----
    #pragma unroll
    for (int i = 0; i < 2; ++i) {
        #pragma unroll
        for (int j = 0; j < 4; ++j) {
            const int mloc = wm * 32 + i * 16 + 8 * half;     // + r
            const int gn   = bn + wn * 64 + j * 16 + l16;
            if (mode == 2) {
                // vT[b,h,d,t]: per lane the 8 accum rows are 8 consecutive t
                const int gm0 = bm + mloc;
                const int b = gm0 >> 10, t0 = gm0 & 1023;
                const int h = gn >> 6,   d  = gn & 63;
                const float bb = bias[gn];
                Pack8 pk;
                #pragma unroll
                for (int r = 0; r < 8; ++r) pk.h[r] = f2bf(acc[i][j][r] + bb);
                ushort_t* dst = (ushort_t*)outp +
                    (((size_t)(b * 16 + h)) * 64 + d) * 1024 + t0;
                *(U128*)dst = pk.u;
            } else if (mode == 3) {
                const float bb = bias[gn];
                float* O = (float*)outp;
                #pragma unroll
                for (int r = 0; r < 8; ++r) {
                    int gm = bm + mloc + r;
                    O[(size_t)gm * 1024 + gn] = acc[i][j][r] + bb;
                }
            } else {
                const float bb = (mode == 0) ? bias[gn] : 0.0f;
                const int h = gn >> 6, d = gn & 63;
                ushort_t* O = (ushort_t*)outp;
                #pragma unroll
                for (int r = 0; r < 8; ++r) {
                    int gm = bm + mloc + r;
                    int b = gm >> 10, t = gm & 1023;
                    O[(((size_t)(b * 16 + h)) * 1024 + t) * 64 + d] =
                        f2bf((acc[i][j][r] + bb) * scale);
                }
            }
        }
    }
}

// ---------------------------------------------------------------------------
// Fused causal attention. Block = (q-tile of 128 rows, bh). 8 waves; each wave
// owns 16 q rows and streams 32-wide k tiles: WMMA scores -> mask -> store qk
// -> online softmax -> P (via per-wave LDS relayout) @ V WMMA accumulate.
// k and vT tiles are staged cooperatively in LDS once per block per k-step.
// ---------------------------------------------------------------------------
__global__ __launch_bounds__(256) void attn_fused(
    const ushort_t* __restrict__ qh, const ushort_t* __restrict__ kh,
    const ushort_t* __restrict__ vT, ushort_t* __restrict__ wv,
    float* __restrict__ qkout) {

    __shared__ ushort_t Ks[32 * 72];      // 32 k-rows x (64+8)  (144B rows, 16B aligned)
    __shared__ ushort_t Vs[64 * 40];      // 64 d-rows x (32+8)
    __shared__ ushort_t Ps[8][16 * 40];   // per-wave P re-layout scratch

    const int qb  = blockIdx.x * 128;
    const int bh  = blockIdx.y;
    const int tid = threadIdx.x;
    const int wid = tid >> 5, lane = tid & 31;
    const int half = lane >> 4, l16 = lane & 15;
    const size_t qkBase = (size_t)bh * 1024 * 1024;

    // q A-fragments (16 rows x hd=64 -> two 16x32 frags), rows are contiguous bf16
    FragAB qf[2];
    {
        const int tqL = qb + wid * 16 + l16;
        const ushort_t* qr = qh + ((size_t)bh * 1024 + tqL) * 64;
        qf[0].u[0] = *(const U128*)&qr[half * 8];
        qf[0].u[1] = *(const U128*)&qr[half * 8 + 16];
        qf[1].u[0] = *(const U128*)&qr[32 + half * 8];
        qf[1].u[1] = *(const U128*)&qr[32 + half * 8 + 16];
    }

    v8f   oacc[4] = {};
    float mrun[8], lrun[8];
    #pragma unroll
    for (int r = 0; r < 8; ++r) { mrun[r] = NEG_INF; lrun[r] = 0.0f; }

    const int tq0  = qb + wid * 16;
    const int kend = (qb + 128 < 1024) ? qb + 128 : 1024;   // causal skip

    const int krow = tid >> 3, kch = (tid & 7) * 8;   // 32 rows x 8 chunks
    const int vrow = tid >> 2, vch = (tid & 3) * 8;   // 64 rows x 4 chunks

    for (int kt = 0; kt < kend; kt += 32) {
        // ---- cooperative LDS staging of k (32x64) and vT (64x32) tiles ----
        stage16(&Ks[krow * 72 + kch],
                kh + ((size_t)bh * 1024 + kt + krow) * 64 + kch);
        stage16(&Vs[vrow * 40 + vch],
                vT + ((size_t)bh * 64 + vrow) * 1024 + kt + vch);
        stage_wait();

        // ---- scores S = q @ k^T for two 16-col subtiles ----
        v8f s[2];
        #pragma unroll
        for (int sub = 0; sub < 2; ++sub) {
            const int kr = (sub * 16 + l16) * 72;
            FragAB k0, k1;
            k0.u[0] = *(const U128*)&Ks[kr + half * 8];
            k0.u[1] = *(const U128*)&Ks[kr + half * 8 + 16];
            k1.u[0] = *(const U128*)&Ks[kr + 32 + half * 8];
            k1.u[1] = *(const U128*)&Ks[kr + 32 + half * 8 + 16];
            v8f z = {};
            z = wmma_bf16(qf[0], k0, z);
            z = wmma_bf16(qf[1], k1, z);
            s[sub] = z;
        }

        // ---- mask, emit qk, online softmax (row = half-wave reduction) ----
        #pragma unroll
        for (int r = 0; r < 8; ++r) {
            const int tq  = tq0 + r + 8 * half;
            const int tk0 = kt + l16, tk1 = kt + 16 + l16;
            float s0 = (tk0 > tq) ? NEG_INF : s[0][r];
            float s1 = (tk1 > tq) ? NEG_INF : s[1][r];
            qkout[qkBase + (size_t)tq * 1024 + tk0] = s0;
            qkout[qkBase + (size_t)tq * 1024 + tk1] = s1;

            float vm = fmaxf(s0, s1);
            #pragma unroll
            for (int off = 1; off < 16; off <<= 1)
                vm = fmaxf(vm, __shfl_xor(vm, off, 32));
            const float newM  = fmaxf(mrun[r], vm);
            const float alpha = __expf(mrun[r] - newM);
            const float e0 = __expf(s0 - newM);
            const float e1 = __expf(s1 - newM);
            float rs = e0 + e1;
            #pragma unroll
            for (int off = 1; off < 16; off <<= 1)
                rs += __shfl_xor(rs, off, 32);
            lrun[r] = lrun[r] * alpha + rs;
            mrun[r] = newM;
            #pragma unroll
            for (int n = 0; n < 4; ++n) oacc[n][r] *= alpha;

            // P in C-layout -> LDS (bf16), for A-fragment re-read
            Ps[wid][(r + 8 * half) * 40 + l16]      = f2bf(e0);
            Ps[wid][(r + 8 * half) * 40 + 16 + l16] = f2bf(e1);
        }

        // ---- P (16x32) A-fragment from LDS, accumulate O += P @ V ----
        FragAB pf;
        pf.u[0] = *(const U128*)&Ps[wid][l16 * 40 + half * 8];
        pf.u[1] = *(const U128*)&Ps[wid][l16 * 40 + half * 8 + 16];
        #pragma unroll
        for (int n = 0; n < 4; ++n) {
            const int vr = (n * 16 + l16) * 40;
            FragAB vf;
            vf.u[0] = *(const U128*)&Vs[vr + half * 8];
            vf.u[1] = *(const U128*)&Vs[vr + half * 8 + 16];
            oacc[n] = wmma_bf16(pf, vf, oacc[n]);
        }
        __syncthreads();   // protect Ks/Vs before next staging pass
    }

    // ---- normalize and emit wv[B,T,D] bf16 ----
    const int b = bh >> 4, h = bh & 15;
    #pragma unroll
    for (int r = 0; r < 8; ++r) {
        const float inv = 1.0f / lrun[r];
        const int tq = tq0 + r + 8 * half;
        const size_t base = ((size_t)b * 1024 + tq) * 1024 + h * 64;
        #pragma unroll
        for (int n = 0; n < 4; ++n)
            wv[base + n * 16 + l16] = f2bf(oacc[n][r] * inv);
    }

    // ---- bulk -inf fill of causally skipped qk region ----
    if (kend < 1024) {
        const int cols = 1024 - kend, q4 = cols >> 2;
        for (int i = tid; i < 128 * q4; i += 256) {
            const int row = i / q4, c = (i % q4) * 4;
            F4 f = {{NEG_INF, NEG_INF, NEG_INF, NEG_INF}};
            *(F4*)&qkout[qkBase + (size_t)(qb + row) * 1024 + kend + c] = f;
        }
    }
}

// ---------------------------------------------------------------------------
extern "C" void kernel_launch(void* const* d_in, const int* in_sizes, int n_in,
                              void* d_out, int out_size, void* d_ws, size_t ws_size,
                              hipStream_t stream) {
    (void)in_sizes; (void)n_in; (void)out_size; (void)ws_size;
    const float* x  = (const float*)d_in[0];
    // d_in[1] = mask: causal, reproduced analytically on-device
    const float* Wq = (const float*)d_in[2];
    const float* bq = (const float*)d_in[3];
    const float* Wk = (const float*)d_in[4];
    const float* Wv = (const float*)d_in[5];
    const float* bv = (const float*)d_in[6];
    const float* Wo = (const float*)d_in[7];
    const float* bo = (const float*)d_in[8];

    const size_t M1 = 1024 * 1024;           // 1M elems
    ushort_t* xb  = (ushort_t*)d_ws;          // [8192,1024] bf16
    ushort_t* wqb = xb  + 8 * M1;
    ushort_t* wkb = wqb + M1;
    ushort_t* wvb = wkb + M1;
    ushort_t* wob = wvb + M1;
    ushort_t* qhB = wob + M1;                 // [B,H,T,hd]
    ushort_t* khB = qhB + 8 * M1;
    ushort_t* vTB = khB + 8 * M1;             // [B,H,hd,T]
    ushort_t* wvB = vTB + 8 * M1;             // [B,T,D]

    float* outF = (float*)d_out;              // out: 8M fp32
    float* qkF  = outF + 8 * M1;              // qk: 128M fp32

    const float scale = 0.35355339059327379f; // 64^-0.25

    cvt_bf16<<<2048, 256, 0, stream>>>(x,  xb,  (int)(8 * M1));
    cvt_bf16<<<1024, 256, 0, stream>>>(Wq, wqb, (int)M1);
    cvt_bf16<<<1024, 256, 0, stream>>>(Wk, wkb, (int)M1);
    cvt_bf16<<<1024, 256, 0, stream>>>(Wv, wvb, (int)M1);
    cvt_bf16<<<1024, 256, 0, stream>>>(Wo, wob, (int)M1);

    dim3 gGemm(8, 64);                        // N/128 x M/128
    gemm_bf16<<<gGemm, 256, 0, stream>>>(xb, wqb, bq, qhB, 0, scale);
    gemm_bf16<<<gGemm, 256, 0, stream>>>(xb, wkb, nullptr, khB, 1, scale);
    gemm_bf16<<<gGemm, 256, 0, stream>>>(xb, wvb, bv, vTB, 2, 1.0f);

    attn_fused<<<dim3(8, 128), 256, 0, stream>>>(qhB, khB, vTB, wvB, qkF);

    gemm_bf16<<<gGemm, 256, 0, stream>>>(wvB, wob, bo, outF, 3, 1.0f);
}
